// PGFMambaBlock_40269613367385
// MI455X (gfx1250) — compile-verified, matrix operand order checked by reading, not run on hardware
//
#include <hip/hip_runtime.h>
#include <hip/hip_bf16.h>
#include <stdint.h>

// ---------------------------------------------------------------------------
// Types for WMMA (gfx1250, wave32)
// ---------------------------------------------------------------------------
typedef __attribute__((ext_vector_type(16))) __bf16 v16bf;
typedef __attribute__((ext_vector_type(8)))  float  v8f;

struct alignas(16) U16x8 { unsigned short u[8]; };
struct alignas(32) FragBF { U16x8 lo, hi; };

static __device__ __forceinline__ unsigned short f2bf(float f) {
    unsigned int u = __float_as_uint(f);
    u += 0x7FFFu + ((u >> 16) & 1u);          // round-to-nearest-even
    return (unsigned short)(u >> 16);
}

static __device__ __forceinline__ float gelu_tanh(float x) {
    float x3 = x * x * x;
    return 0.5f * x * (1.0f + tanhf(0.7978845608028654f * (x + 0.044715f * x3)));
}

static __device__ __forceinline__ float softplus_f(float x) {
    return (x > 20.0f) ? x : log1pf(expf(x));
}

static __device__ __forceinline__ float phi1_f(float z) {
    if (fabsf(z) < 1e-4f) return 1.0f + 0.5f * z;
    return expm1f(z) / z;
}

// ---------------------------------------------------------------------------
// gfx1250 async memory->LDS copy (ASYNCcnt tracked), 16B per lane.
// LDS destination address = low 32 bits of the generic pointer
// ---------------------------------------------------------------------------
static __device__ __forceinline__ void async_copy_b128(void* lds_ptr, const void* gptr) {
    unsigned lds_addr = (unsigned)(size_t)lds_ptr;
    unsigned long long ga = (unsigned long long)(size_t)gptr;
    asm volatile("global_load_async_to_lds_b128 %0, %1, off"
                 :: "v"(lds_addr), "v"(ga)
                 : "memory");
}

template <int Nc>
static __device__ __forceinline__ void wait_asynccnt() {
    asm volatile("s_wait_asynccnt %0" :: "n"(Nc) : "memory");
}

// ---------------------------------------------------------------------------
// f32 -> bf16 convert
// ---------------------------------------------------------------------------
__global__ void cvt_bf16_kernel(const float* __restrict__ in,
                                unsigned short* __restrict__ out, int n) {
    int i = blockIdx.x * 256 + threadIdx.x;
    if (i < n) out[i] = f2bf(in[i]);
}

// ---------------------------------------------------------------------------
// RMSNorm: one block per row of length Dn=1024; writes f32 (optional) + bf16
// ---------------------------------------------------------------------------
template <int WRITE_F32>
__global__ __launch_bounds__(256)
void rmsnorm_kernel(const float* __restrict__ x, const float* __restrict__ w,
                    float* __restrict__ xf, unsigned short* __restrict__ xb, int Dn) {
    const int row = blockIdx.x;
    const int tid = threadIdx.x;
    const float* xr = x + (size_t)row * Dn;
    float ss = 0.0f;
    for (int i = tid; i < Dn; i += 256) { float v = xr[i]; ss += v * v; }
    __shared__ float red[256];
    red[tid] = ss;
    __syncthreads();
    for (int s = 128; s > 0; s >>= 1) {
        if (tid < s) red[tid] += red[tid + s];
        __syncthreads();
    }
    const float r = rsqrtf(red[0] / (float)Dn + 1e-6f);
    for (int i = tid; i < Dn; i += 256) {
        float v = xr[i] * r * w[i];
        if (WRITE_F32) xf[(size_t)row * Dn + i] = v;
        xb[(size_t)row * Dn + i] = f2bf(v);
    }
}

// ---------------------------------------------------------------------------
// WMMA bf16 GEMM:  out[m][n] = sum_k A[m][k] * Bw[n][k]  (+epilogue)
// MODE 0: outF = acc + bias[n]                      (f32)
// MODE 1: outB = bf16(gelu(acc + bias[n]))          (bf16)
// MODE 2: outF = acc + bias[n] + resid[m][n]        (f32)
// Block: 256 threads = 8 waves (2x4), block tile 128x128, BK=64.
// Wave tile 64x32 = 4x2 accumulators of v_wmma_f32_16x16x32_bf16.
// Double-buffered LDS, filled by global_load_async_to_lds_b128 (ASYNCcnt).
// ---------------------------------------------------------------------------
template <int MODE>
__global__ __launch_bounds__(256)
void gemm_wmma_bf16(const unsigned short* __restrict__ A,
                    const unsigned short* __restrict__ Bw,
                    const float* __restrict__ bias,
                    const float* __restrict__ resid,
                    float* __restrict__ outF,
                    unsigned short* __restrict__ outB,
                    int M, int N, int K) {
    __shared__ unsigned short As[2][128][72];  // 64 cols + pad, 16B-aligned rows
    __shared__ unsigned short Bs[2][128][72];

    const int tid   = threadIdx.x;
    const int lane  = tid & 31;
    const int wid   = tid >> 5;
    const int waveM = wid >> 2;      // 0..1  (rows of 64)
    const int waveN = wid & 3;       // 0..3  (cols of 32)
    const int m0 = blockIdx.y * 128;
    const int n0 = blockIdx.x * 128;

    const int lrow = tid & 127;      // LDS fill: row within tile
    const int lch  = tid >> 7;       // 0..1 : which 32-col half

    const int ml   = lane & 15;      // M (or N) index within fragment
    const int half = lane >> 4;      // lane-half selects K sub-pattern

    const v8f vzero = {0.f, 0.f, 0.f, 0.f, 0.f, 0.f, 0.f, 0.f};
    v8f acc[4][2];
#pragma unroll
    for (int i = 0; i < 4; ++i)
#pragma unroll
        for (int j = 0; j < 2; ++j) acc[i][j] = vzero;

    const unsigned short* rowA = A  + (size_t)(m0 + lrow) * K + lch * 32;
    const unsigned short* rowB = Bw + (size_t)(n0 + lrow) * K + lch * 32;

    // Issue one K-slab (8 x 16B per thread) into LDS buffer `buf` via async DMA.
    auto issue_slab = [&](int buf, int k0) {
        const unsigned short* sA = rowA + k0;
        const unsigned short* sB = rowB + k0;
        unsigned short* dA = &As[buf][lrow][lch * 32];
        unsigned short* dB = &Bs[buf][lrow][lch * 32];
        async_copy_b128(dA + 0,  sA + 0);
        async_copy_b128(dA + 8,  sA + 8);
        async_copy_b128(dA + 16, sA + 16);
        async_copy_b128(dA + 24, sA + 24);
        async_copy_b128(dB + 0,  sB + 0);
        async_copy_b128(dB + 8,  sB + 8);
        async_copy_b128(dB + 16, sB + 16);
        async_copy_b128(dB + 24, sB + 24);
    };

    const int nslab = K >> 6;
    issue_slab(0, 0);

    for (int i = 0; i < nslab; ++i) {
        const int cur = i & 1;
        if (i + 1 < nslab) {
            issue_slab(cur ^ 1, (i + 1) << 6);  // prefetch next slab (async)
            wait_asynccnt<8>();                 // this wave's slab-i loads done
        } else {
            wait_asynccnt<0>();
        }
        __syncthreads();                        // all waves' slab-i data in LDS

#pragma unroll
        for (int s = 0; s < 2; ++s) {
            const int ks = s * 32;
            // 16-bit A/B fragment layout: lane-half 0 holds K {0..7,16..23},
            // half 1 holds K {8..15,24..31} (two 16B runs per lane).
            FragBF fa[4], fb[2];
#pragma unroll
            for (int ii = 0; ii < 4; ++ii) {
                const int row = waveM * 64 + ii * 16 + ml;
                fa[ii].lo = *(const U16x8*)&As[cur][row][ks + half * 8];
                fa[ii].hi = *(const U16x8*)&As[cur][row][ks + 16 + half * 8];
            }
#pragma unroll
            for (int j = 0; j < 2; ++j) {
                const int col = waveN * 32 + j * 16 + ml;
                fb[j].lo = *(const U16x8*)&Bs[cur][col][ks + half * 8];
                fb[j].hi = *(const U16x8*)&Bs[cur][col][ks + 16 + half * 8];
            }
#pragma unroll
            for (int ii = 0; ii < 4; ++ii) {
                v16bf av = __builtin_bit_cast(v16bf, fa[ii]);
#pragma unroll
                for (int j = 0; j < 2; ++j) {
                    v16bf bv = __builtin_bit_cast(v16bf, fb[j]);
                    acc[ii][j] = __builtin_amdgcn_wmma_f32_16x16x32_bf16(
                        false, av, false, bv, (short)0, acc[ii][j], false, false);
                }
            }
        }
        __syncthreads();   // everyone finished reading buf `cur` before refill
    }

    // Epilogue: C/D layout — VGPR r: lanes 0-15 -> M=r, lanes 16-31 -> M=r+8.
#pragma unroll
    for (int i = 0; i < 4; ++i) {
        const int rowBase = m0 + waveM * 64 + i * 16;
#pragma unroll
        for (int j = 0; j < 2; ++j) {
            const int col = n0 + waveN * 32 + j * 16 + ml;
#pragma unroll
            for (int r = 0; r < 8; ++r) {
                const int row = rowBase + r + half * 8;
                float c = acc[i][j][r] + bias[col];
                const size_t o = (size_t)row * N + col;
                if (MODE == 0) {
                    outF[o] = c;
                } else if (MODE == 1) {
                    outB[o] = f2bf(gelu_tanh(c));
                } else {
                    outF[o] = c + resid[o];
                }
            }
        }
    }
}

// ---------------------------------------------------------------------------
// Bv / Cv projections: out[row][n] = dot(xn[row,:], W[n,:]) + b[n]
// ---------------------------------------------------------------------------
__global__ __launch_bounds__(256)
void bc_proj_kernel(const float* __restrict__ xnf,
                    const float* __restrict__ WB, const float* __restrict__ bB,
                    const float* __restrict__ WC, const float* __restrict__ bC,
                    float* __restrict__ Bv, float* __restrict__ Cv, int K) {
    const int t = blockIdx.x * 256 + threadIdx.x;
    const int row = t >> 5;
    const int o = t & 31;
    const int n = o & 15;
    const bool isC = o >= 16;
    const float* w = (isC ? WC : WB) + (size_t)n * K;
    const float* xr = xnf + (size_t)row * K;
    float s = 0.0f;
    for (int k = 0; k < K; k += 4) {
        s += xr[k] * w[k] + xr[k + 1] * w[k + 1]
           + xr[k + 2] * w[k + 2] + xr[k + 3] * w[k + 3];
    }
    s += (isC ? bC : bB)[n];
    (isC ? Cv : Bv)[(size_t)row * 16 + n] = s;
}

// ---------------------------------------------------------------------------
// Chunked linear-recurrence scan (h = dA*h + b), 3 passes, NC chunks of CL.
// Thread layout (pass 1/3): 16 lanes (n) per (b,d,chunk); wave32 = 2 groups.
// ---------------------------------------------------------------------------
#define SCAN_NC 32
#define SCAN_CL 64

// Pass 1: per-chunk transfer (P = prod dA, q = end state from h0=0)
__global__ __launch_bounds__(256)
void scan_pass1(const float* __restrict__ dtp, const float* __restrict__ xnf,
                const float* __restrict__ Bv, const float* __restrict__ A_log,
                float* __restrict__ Pbuf, float* __restrict__ Qbuf,
                int Lv, int Dv) {
    const int tid = blockIdx.x * 256 + threadIdx.x;
    const int n  = tid & 15;
    const int pc = tid >> 4;              // (b,d,chunk)
    const int c  = pc & (SCAN_NC - 1);
    const int p  = pc >> 5;               // log2(SCAN_NC)=5
    const int b  = p / Dv;
    const int d  = p - b * Dv;

    const float An = -expf(A_log[n]);
    float h = 0.0f, pf = 1.0f;
    const int l0 = c * SCAN_CL;
    for (int l = l0; l < l0 + SCAN_CL; ++l) {
        const size_t idx = (size_t)(b * Lv + l);
        const float dt  = softplus_f(dtp[idx * Dv + d]);
        const float xv  = xnf[idx * Dv + d];
        const float dtA = dt * An;
        const float dA  = expf(dtA);
        const float bt  = phi1_f(dtA) * Bv[idx * 16 + n] * xv;
        h  = dA * h + bt;
        pf = dA * pf;
    }
    Pbuf[(size_t)pc * 16 + n] = pf;
    Qbuf[(size_t)pc * 16 + n] = h;
}

// Pass 2: sequential combine over chunks -> initial state H_c per chunk
__global__ __launch_bounds__(256)
void scan_pass2(const float* __restrict__ Pbuf, const float* __restrict__ Qbuf,
                float* __restrict__ Hbuf) {
    const int tid = blockIdx.x * 256 + threadIdx.x;
    const int n = tid & 15;
    const int p = tid >> 4;               // (b,d)
    float H = 0.0f;
    for (int c = 0; c < SCAN_NC; ++c) {
        const size_t o = ((size_t)p * SCAN_NC + c) * 16 + n;
        Hbuf[o] = H;
        H = Pbuf[o] * H + Qbuf[o];
    }
}

// Pass 3: replay chunk from true H_c, emit y = sum_n Cv*h + D*xn, +residual
__global__ __launch_bounds__(256)
void scan_pass3(const float* __restrict__ dtp, const float* __restrict__ xnf,
                const float* __restrict__ xres,
                const float* __restrict__ Bv, const float* __restrict__ Cv,
                const float* __restrict__ A_log, const float* __restrict__ Dp,
                const float* __restrict__ scale, const float* __restrict__ Hbuf,
                float* __restrict__ x2, int Lv, int Dv) {
    const int tid = blockIdx.x * 256 + threadIdx.x;
    const int n  = tid & 15;
    const int pc = tid >> 4;
    const int c  = pc & (SCAN_NC - 1);
    const int p  = pc >> 5;
    const int b  = p / Dv;
    const int d  = p - b * Dv;

    const float An = -expf(A_log[n]);
    const float Dd = Dp[d];
    const float sc = scale[d];

    float h = Hbuf[(size_t)pc * 16 + n];
    const int l0 = c * SCAN_CL;
    for (int l = l0; l < l0 + SCAN_CL; ++l) {
        const size_t idx = (size_t)(b * Lv + l);
        const float dt  = softplus_f(dtp[idx * Dv + d]);
        const float xv  = xnf[idx * Dv + d];
        const float dtA = dt * An;
        const float dA  = expf(dtA);
        const float bt  = phi1_f(dtA) * Bv[idx * 16 + n] * xv;
        h = dA * h + bt;
        float contrib = Cv[idx * 16 + n] * h;
        contrib += __shfl_xor(contrib, 8, 16);
        contrib += __shfl_xor(contrib, 4, 16);
        contrib += __shfl_xor(contrib, 2, 16);
        contrib += __shfl_xor(contrib, 1, 16);
        if (n == 0) {
            float y = (contrib + Dd * xv) * sc;
            x2[idx * Dv + d] = y + xres[idx * Dv + d];
        }
    }
}

// ---------------------------------------------------------------------------
// Launch
// ---------------------------------------------------------------------------
extern "C" void kernel_launch(void* const* d_in, const int* in_sizes, int n_in,
                              void* d_out, int out_size, void* d_ws, size_t ws_size,
                              hipStream_t stream) {
    (void)in_sizes; (void)n_in; (void)out_size; (void)ws_size;
    constexpr int Bb = 2, Lv = 2048, Dv = 1024, Nst = 16, DFF = 4096;
    constexpr int Mrows = Bb * Lv;  // 4096
    constexpr int NCh = SCAN_NC;

    const float* x       = (const float*)d_in[0];
    const float* norm1_w = (const float*)d_in[1];
    const float* norm2_w = (const float*)d_in[2];
    const float* A_log   = (const float*)d_in[3];
    const float* D_param = (const float*)d_in[4];
    const float* scale   = (const float*)d_in[5];
    const float* Wdt     = (const float*)d_in[6];
    const float* bdt     = (const float*)d_in[7];
    const float* WB      = (const float*)d_in[8];
    const float* bB      = (const float*)d_in[9];
    const float* WC      = (const float*)d_in[10];
    const float* bC      = (const float*)d_in[11];
    const float* W1      = (const float*)d_in[12];
    const float* b1      = (const float*)d_in[13];
    const float* W2      = (const float*)d_in[14];
    const float* b2      = (const float*)d_in[15];
    float* out = (float*)d_out;

    size_t off = 0;
    auto alloc = [&](size_t bytes) -> void* {
        void* p = (char*)d_ws + off;
        off += (bytes + 255) & ~(size_t)255;
        return p;
    };
    float*          xnf  = (float*)alloc((size_t)Mrows * Dv * 4);
    unsigned short* xnb  = (unsigned short*)alloc((size_t)Mrows * Dv * 2);
    unsigned short* wdtb = (unsigned short*)alloc((size_t)Dv * Dv * 2);
    unsigned short* w1b  = (unsigned short*)alloc((size_t)DFF * Dv * 2);
    unsigned short* w2b  = (unsigned short*)alloc((size_t)Dv * DFF * 2);
    float*          dtp  = (float*)alloc((size_t)Mrows * Dv * 4);
    float*          Bv   = (float*)alloc((size_t)Mrows * Nst * 4);
    float*          Cv   = (float*)alloc((size_t)Mrows * Nst * 4);
    float*          x2   = (float*)alloc((size_t)Mrows * Dv * 4);
    unsigned short* xn2b = (unsigned short*)alloc((size_t)Mrows * Dv * 2);
    unsigned short* hb   = (unsigned short*)alloc((size_t)Mrows * DFF * 2);
    float*          Pbuf = (float*)alloc((size_t)Bb * Dv * NCh * 16 * 4);
    float*          Qbuf = (float*)alloc((size_t)Bb * Dv * NCh * 16 * 4);
    float*          Hbuf = (float*)alloc((size_t)Bb * Dv * NCh * 16 * 4);

    // 1. weight converts to bf16
    cvt_bf16_kernel<<<(Dv * Dv + 255) / 256, 256, 0, stream>>>(Wdt, wdtb, Dv * Dv);
    cvt_bf16_kernel<<<(DFF * Dv + 255) / 256, 256, 0, stream>>>(W1, w1b, DFF * Dv);
    cvt_bf16_kernel<<<(Dv * DFF + 255) / 256, 256, 0, stream>>>(W2, w2b, Dv * DFF);

    // 2. RMSNorm1 (f32 + bf16)
    rmsnorm_kernel<1><<<Mrows, 256, 0, stream>>>(x, norm1_w, xnf, xnb, Dv);

    // 3. dt_pre = xn @ Wdt^T + bdt  (WMMA, async-LDS pipelined)
    gemm_wmma_bf16<0><<<dim3(Dv / 128, Mrows / 128), 256, 0, stream>>>(
        xnb, wdtb, bdt, nullptr, dtp, nullptr, Mrows, Dv, Dv);

    // 4. Bv / Cv projections
    bc_proj_kernel<<<(Mrows * 32) / 256, 256, 0, stream>>>(
        xnf, WB, bB, WC, bC, Bv, Cv, Dv);

    // 5. chunked selective scan + residual -> x2
    scan_pass1<<<(Bb * Dv * NCh * 16) / 256, 256, 0, stream>>>(
        dtp, xnf, Bv, A_log, Pbuf, Qbuf, Lv, Dv);
    scan_pass2<<<(Bb * Dv * 16) / 256, 256, 0, stream>>>(Pbuf, Qbuf, Hbuf);
    scan_pass3<<<(Bb * Dv * NCh * 16) / 256, 256, 0, stream>>>(
        dtp, xnf, x, Bv, Cv, A_log, D_param, scale, Hbuf, x2, Lv, Dv);

    // 6. RMSNorm2 (bf16 only)
    rmsnorm_kernel<0><<<Mrows, 256, 0, stream>>>(x2, norm2_w, nullptr, xn2b, Dv);

    // 7. h = gelu(xn2 @ W1^T + b1) -> bf16  (WMMA)
    gemm_wmma_bf16<1><<<dim3(DFF / 128, Mrows / 128), 256, 0, stream>>>(
        xn2b, w1b, b1, nullptr, nullptr, hb, Mrows, DFF, Dv);

    // 8. out = h @ W2^T + b2 + x2  (WMMA)
    gemm_wmma_bf16<2><<<dim3(Dv / 128, Mrows / 128), 256, 0, stream>>>(
        hb, w2b, b2, x2, out, nullptr, Mrows, Dv, DFF);
}
